// LocalModule_49572512530878
// MI455X (gfx1250) — matrix operand order repaired and checked by reading, not compiled, not run on hardware
//
#include <hip/hip_runtime.h>
#include <hip/hip_bf16.h>

typedef __attribute__((ext_vector_type(16))) __bf16 v16bf;
typedef __attribute__((ext_vector_type(8)))  float  v8f;

#define VIEWS 32
#define KNN 5
#define DD 1024
#define BATCH 512
#define M_TOTAL (BATCH * VIEWS)   // 16384
#define IMG_STRIDE (3 * 34 * 34)  // 3468
#define BROW_PITCH 80             // 32 bf16 (64B) + 16B pad -> conflict-free b128 LDS reads

// ---------------------------------------------------------------------------
// CDNA5 async-tensor helpers (inline asm: portable across both toolchains).
// global_load_async_to_lds_b128: VDST = per-lane LDS byte offset, VADDR =
// per-lane 64-bit global address, tracked by ASYNCcnt.
// ---------------------------------------------------------------------------
__device__ __forceinline__ void async_b128_to_lds(unsigned lds_off, const void* gptr) {
    asm volatile("global_load_async_to_lds_b128 %0, %1, off"
                 :: "v"(lds_off), "v"(gptr) : "memory");
}
__device__ __forceinline__ void wait_asynccnt0() {
    asm volatile("s_wait_asynccnt 0x0" ::: "memory");
}

// ---------------------------------------------------------------------------
// Kernel 1: kNN over the 32 vertices + relation MLP. Batch-invariant, so we
// compute the (32 x 5) neighbor indices and edge weights exactly once.
// ---------------------------------------------------------------------------
__global__ void relation_kernel(const float* __restrict__ vert,
                                const float* __restrict__ r_w1,
                                const float* __restrict__ r_b1,
                                const float* __restrict__ r_w2,
                                const float* __restrict__ r_b2,
                                float* __restrict__ e_out,
                                int* __restrict__ idx_out) {
    int n = threadIdx.x;
    if (n >= VIEWS) return;
    float vx = vert[n * 3 + 0], vy = vert[n * 3 + 1], vz = vert[n * 3 + 2];
    float pd[VIEWS];
    for (int m = 0; m < VIEWS; ++m) {
        float dx = vx - vert[m * 3 + 0];
        float dy = vy - vert[m * 3 + 1];
        float dz = vz - vert[m * 3 + 2];
        pd[m] = -(dx * dx + dy * dy + dz * dz);   // negative squared distance
    }
    int  sel[KNN];
    bool used[VIEWS];
    for (int m = 0; m < VIEWS; ++m) used[m] = false;
    for (int k = 0; k < KNN; ++k) {
        int best = 0; float bv = -3.0e38f;
        for (int m = 0; m < VIEWS; ++m)
            if (!used[m] && pd[m] > bv) { bv = pd[m]; best = m; }
        used[best] = true;
        sel[k] = best;
    }
    float v0x = vert[sel[0] * 3 + 0];
    float v0y = vert[sel[0] * 3 + 1];
    float v0z = vert[sel[0] * 3 + 2];
    for (int k = 0; k < KNN; ++k) {
        float ux = vert[sel[k] * 3 + 0];
        float uy = vert[sel[k] * 3 + 1];
        float uz = vert[sel[k] * 3 + 2];
        float dx = v0x - ux, dy = v0y - uy, dz = v0z - uz;
        float dn = sqrtf(dx * dx + dy * dy + dz * dz + 1e-12f);
        float vf[10] = {v0x, v0y, v0z, ux, uy, uz, dx, dy, dz, dn};
        float h[10];
        #pragma unroll
        for (int j = 0; j < 10; ++j) {
            float s = r_b1[j];
            #pragma unroll
            for (int i = 0; i < 10; ++i) s += vf[i] * r_w1[j * 10 + i];
            h[j] = s > 0.f ? s : 0.f;
        }
        float ev = r_b2[0];
        #pragma unroll
        for (int j = 0; j < 10; ++j) ev += h[j] * r_w2[j];
        e_out[n * KNN + k]  = ev;
        idx_out[n * KNN + k] = sel[k];
    }
}

// ---------------------------------------------------------------------------
// Kernel 2: cast lin_w (1024x1024 f32, row-major, K contiguous) to bf16.
// ---------------------------------------------------------------------------
__global__ void convw_kernel(const float* __restrict__ w,
                             __bf16* __restrict__ wb, int nelem) {
    int i = blockIdx.x * blockDim.x + threadIdx.x;
    if (i < nelem) wb[i] = (__bf16)w[i];
}

// ---------------------------------------------------------------------------
// Kernel 3: fused Conv2d(3->1,k3,valid) + view-graph gather.
// One block per batch element: all 32 views' conv output (32x1024 f32 =
// 128KB) is staged in LDS (320KB/WGP), then the K=5 weighted gather runs
// in-LDS and emits the bf16 GEMM activation matrix directly.
// ---------------------------------------------------------------------------
__global__ void __launch_bounds__(1024)
conv_gather_kernel(const float* __restrict__ x,
                   const float* __restrict__ cw,
                   const float* __restrict__ cb,
                   const float* __restrict__ e,
                   const int* __restrict__ idx,
                   __bf16* __restrict__ Abf) {
    extern __shared__ float smem[];
    float* Fs = smem;                              // 32*1024 floats
    float* es = smem + VIEWS * DD;                 // 160 floats
    int*   is = (int*)(es + VIEWS * KNN);          // 160 ints

    const int b = blockIdx.x;
    const int t = threadIdx.x;                     // 0..1023

    if (t < VIEWS * KNN) { es[t] = e[t]; is[t] = idx[t]; }

    float w[27];
    #pragma unroll
    for (int i = 0; i < 27; ++i) w[i] = cw[i];
    const float bias = cb[0];

    for (int i = t; i < VIEWS * DD; i += 1024) {
        const int v  = i >> 10;
        const int p  = i & 1023;
        const int oy = p >> 5, ox = p & 31;
        const float* xi = x + (size_t)(b * VIEWS + v) * IMG_STRIDE;
        float acc = bias;
        #pragma unroll
        for (int c = 0; c < 3; ++c)
            #pragma unroll
            for (int ky = 0; ky < 3; ++ky)
                #pragma unroll
                for (int kx = 0; kx < 3; ++kx)
                    acc += xi[c * 1156 + (oy + ky) * 34 + (ox + kx)] * w[c * 9 + ky * 3 + kx];
        Fs[i] = acc;
    }
    __syncthreads();
    for (int i = t; i < VIEWS * DD; i += 1024) {
        const int n = i >> 10, d = i & 1023;
        float acc = 0.f;
        #pragma unroll
        for (int k = 0; k < KNN; ++k)
            acc += es[n * KNN + k] * Fs[(is[n * KNN + k] << 10) + d];
        Abf[((size_t)(b * VIEWS + n) << 10) + d] = (__bf16)acc;
    }
}

// ---------------------------------------------------------------------------
// Kernel 4: Z = A(16384x1024 bf16) * W(1024x1024 bf16, N x K)^T + lin_b,
// f32 accumulate via V_WMMA_F32_16X16X32_BF16.
// Block = 256 threads = 8 waves, 128x128 output tile; each wave owns a
// 32(M) x 64(N) sub-tile = 2x4 WMMA accumulators.
// The shared 32x128 B tile is staged into LDS with CDNA5 async loads
// (GLOBAL_LOAD_ASYNC_TO_LDS_B128, ASYNCcnt-tracked), double buffered so
// staging of K-step k+1 overlaps the WMMA stream of K-step k. A fragments
// are fetched directly (only 2x reuse; keeps vmem + ds pipes both busy).
// ---------------------------------------------------------------------------
__global__ void __launch_bounds__(256)
gemm_kernel(const __bf16* __restrict__ A,
            const __bf16* __restrict__ W,
            const float* __restrict__ lin_b,
            float* __restrict__ Z) {
    __shared__ __align__(16) char Bs[2][128 * BROW_PITCH];   // 20 KB

    const int lane = threadIdx.x & 31;
    const int wave = threadIdx.x >> 5;
    const int wm = wave & 3;           // 4 waves along M
    const int wn = wave >> 2;          // 2 waves along N
    const int Mbase  = blockIdx.x * 128 + wm * 32;
    const int NbaseG = blockIdx.y * 128;            // block's N origin
    const int nloc0  = wn * 64;                     // wave's N offset in tile
    const int l16 = lane & 15;
    const int lhi = lane >> 4;

    // Async-stage one 32x128 B K-slab: 512 x 16B chunks, 2 per thread.
    auto issueB = [&](int k0, int buf) {
        #pragma unroll
        for (int r = 0; r < 2; ++r) {
            const int c = (int)threadIdx.x + r * 256;   // 0..511
            const int n = c >> 2;                       // row in tile (N)
            const int q = c & 3;                        // 16B chunk in row
            unsigned lds_off = (unsigned)(uintptr_t)&Bs[buf][n * BROW_PITCH + q * 16];
            const void* g = (const void*)(W + (size_t)(NbaseG + n) * DD + k0 + q * 8);
            async_b128_to_lds(lds_off, g);
        }
    };

    v8f acc[2][4];
    #pragma unroll
    for (int i = 0; i < 2; ++i)
        #pragma unroll
        for (int j = 0; j < 4; ++j) {
            v8f z = {};
            acc[i][j] = z;
        }

    issueB(0, 0);   // prologue

    for (int k0 = 0; k0 < DD; k0 += 32) {
        const int cur = (k0 >> 5) & 1;
        wait_asynccnt0();      // our async writes into Bs[cur] have landed
        __syncthreads();       // everyone's have; prior readers of Bs[cur^1] done
        if (k0 + 32 < DD) issueB(k0 + 32, cur ^ 1);

        // A fragments (direct): lane<16 holds M=l16, K {0..7,16..23};
        //                       lane>=16 holds K {8..15,24..31}
        v16bf afrag[2];
        #pragma unroll
        for (int mt = 0; mt < 2; ++mt) {
            const uint4* pa = (const uint4*)(A + (size_t)(Mbase + mt * 16 + l16) * DD
                                               + k0 + (lhi << 3));
            union { uint4 u[2]; v16bf v; } ua;
            ua.u[0] = pa[0];
            ua.u[1] = pa[2];
            afrag[mt] = ua.v;
        }
        // B fragments from LDS: lane<16 holds N=l16, K {0..7,16..23}; lane>=16 K {8..15,24..31}
        v16bf bfrag[4];
        #pragma unroll
        for (int nt = 0; nt < 4; ++nt) {
            const uint4* pb = (const uint4*)&Bs[cur][(nloc0 + nt * 16 + l16) * BROW_PITCH
                                                     + lhi * 16];
            union { uint4 u[2]; v16bf v; } ub;
            ub.u[0] = pb[0];
            ub.u[1] = pb[2];
            bfrag[nt] = ub.v;
        }
        #pragma unroll
        for (int mt = 0; mt < 2; ++mt)
            #pragma unroll
            for (int nt = 0; nt < 4; ++nt)
                acc[mt][nt] = __builtin_amdgcn_wmma_f32_16x16x32_bf16(
                    false, afrag[mt], false, bfrag[nt],
                    (short)0, acc[mt][nt], false, false);
    }

    // Epilogue: C/D layout — lane<16: N=lane, M=r; lane>=16: N=lane-16, M=r+8
    #pragma unroll
    for (int mt = 0; mt < 2; ++mt) {
        const int rowb = Mbase + mt * 16 + lhi * 8;
        #pragma unroll
        for (int nt = 0; nt < 4; ++nt) {
            const int col = NbaseG + nloc0 + nt * 16 + l16;
            const float bias = lin_b[col];
            #pragma unroll
            for (int r = 0; r < 8; ++r)
                Z[(size_t)(rowb + r) * DD + col] = acc[mt][nt][r] + bias;
        }
    }
}

// ---------------------------------------------------------------------------
// Kernel 5: per-column sum / sum-of-squares over the 16384 rows (coalesced:
// adjacent lanes read adjacent columns), atomically accumulated.
// ---------------------------------------------------------------------------
__global__ void __launch_bounds__(256)
colstats_kernel(const float* __restrict__ Z,
                float* __restrict__ cs, float* __restrict__ cs2) {
    const int t  = threadIdx.x;   // 256
    const int rb = blockIdx.x;    // 64 blocks x 256 rows
    #pragma unroll
    for (int cb = 0; cb < 4; ++cb) {
        const int col = cb * 256 + t;
        float s = 0.f, s2 = 0.f;
        const float* p = Z + (size_t)rb * 256 * DD + col;
        for (int r = 0; r < 256; ++r) {
            float v = p[(size_t)r * DD];
            s += v; s2 += v * v;
        }
        atomicAdd(&cs[col], s);
        atomicAdd(&cs2[col], s2);
    }
}

// ---------------------------------------------------------------------------
// Kernel 6: BatchNorm (mean/var over rows) + LeakyReLU(0.2) -> d_out (f32).
// ---------------------------------------------------------------------------
__global__ void __launch_bounds__(256)
bn_kernel(const float* __restrict__ Z,
          const float* __restrict__ cs, const float* __restrict__ cs2,
          const float* __restrict__ g, const float* __restrict__ bb,
          float* __restrict__ out) {
    const size_t i = (size_t)blockIdx.x * 256 + threadIdx.x;
    const int col = (int)(i & (DD - 1));
    const float invN = 1.0f / (float)M_TOTAL;
    const float mean = cs[col] * invN;
    const float var  = cs2[col] * invN - mean * mean;
    const float inv  = rsqrtf(var + 1e-5f);
    const float zv   = (Z[i] - mean) * inv * g[col] + bb[col];
    out[i] = zv >= 0.f ? zv : 0.2f * zv;
}

// ---------------------------------------------------------------------------
extern "C" void kernel_launch(void* const* d_in, const int* in_sizes, int n_in,
                              void* d_out, int out_size, void* d_ws, size_t ws_size,
                              hipStream_t stream) {
    const float* x      = (const float*)d_in[0];
    const float* vert   = (const float*)d_in[1];
    const float* conv_w = (const float*)d_in[2];
    const float* conv_b = (const float*)d_in[3];
    const float* r_w1   = (const float*)d_in[4];
    const float* r_b1   = (const float*)d_in[5];
    const float* r_w2   = (const float*)d_in[6];
    const float* r_b2   = (const float*)d_in[7];
    const float* lin_w  = (const float*)d_in[8];
    const float* lin_b  = (const float*)d_in[9];
    const float* bn_g   = (const float*)d_in[10];
    const float* bn_b   = (const float*)d_in[11];
    float* out = (float*)d_out;

    // Workspace layout:
    char* ws = (char*)d_ws;
    size_t off = 0;
    __bf16* Abf = (__bf16*)(ws + off); off += (size_t)M_TOTAL * DD * 2;  // 32 MB
    __bf16* Wbf = (__bf16*)(ws + off); off += (size_t)DD * DD * 2;       //  2 MB
    float*  Z   = (float*)(ws + off);  off += (size_t)M_TOTAL * DD * 4;  // 64 MB
    float*  e_buf   = (float*)(ws + off); off += 256 * 4;
    int*    idx_buf = (int*)(ws + off);   off += 256 * 4;
    float*  cs  = (float*)(ws + off);  off += DD * 4;                    // colsum
    float*  cs2 = (float*)(ws + off);  off += DD * 4;                    // colsumsq

    // zero the atomic accumulators (cs, cs2 contiguous) — graph-capturable
    hipMemsetAsync(cs, 0, 2 * DD * sizeof(float), stream);

    relation_kernel<<<1, 32, 0, stream>>>(vert, r_w1, r_b1, r_w2, r_b2,
                                          e_buf, idx_buf);

    convw_kernel<<<(DD * DD + 255) / 256, 256, 0, stream>>>(lin_w, Wbf, DD * DD);

    const size_t smem = (size_t)VIEWS * DD * 4 + VIEWS * KNN * 4 + VIEWS * KNN * 4;
    conv_gather_kernel<<<BATCH, 1024, smem, stream>>>(x, conv_w, conv_b,
                                                      e_buf, idx_buf, Abf);

    dim3 gg(M_TOTAL / 128, DD / 128);   // (128, 8)
    gemm_kernel<<<gg, 256, 0, stream>>>(Abf, Wbf, lin_b, Z);

    colstats_kernel<<<64, 256, 0, stream>>>(Z, cs, cs2);

    bn_kernel<<<(M_TOTAL * DD) / 256, 256, 0, stream>>>(Z, cs, cs2, bn_g, bn_b, out);
}
